// FWLNet_34857954574875
// MI455X (gfx1250) — matrix operand order repaired and checked by reading, not compiled, not run on hardware
//
#include <hip/hip_runtime.h>
#include <stdint.h>

#define N_NODES 1024
#define NN      (1024u * 1024u)
#define NE      16384
#define NQ      8192
#define H       32
#define EPSV    1e-5f

typedef __attribute__((ext_vector_type(16))) __bf16 bf16x16;
typedef __attribute__((ext_vector_type(8)))  float  f32x8;
typedef int v4i __attribute__((vector_size(16)));

union FragAB { bf16x16 v; uint4 q[2]; };

__device__ __forceinline__ uint16_t f2bf(float f) {
  union { float f; uint32_t u; } v; v.f = f;
  uint32_t r = v.u + 0x7fffu + ((v.u >> 16) & 1u);   // round-to-nearest-even
  return (uint16_t)(r >> 16);
}

// CDNA5 async global->LDS (ASYNCcnt path); fall back to sync staging if the
// toolchain doesn't declare the builtins.
#if __has_builtin(__builtin_amdgcn_global_load_async_to_lds_b128) && \
    __has_builtin(__builtin_amdgcn_s_wait_asynccnt)
#define HAS_ASYNC_LDS 1
#else
#define HAS_ASYNC_LDS 0
#endif

__device__ __forceinline__ void stage_chunk(const uint16_t* __restrict__ g,
                                            uint16_t* __restrict__ l) {
#if HAS_ASYNC_LDS
  // builtin expects (global int4*, lds int4*, imm offset, imm cpol);
  // generic LDS pointer -> 32-bit LDS offset (low 32 bits per aperture rules)
  __builtin_amdgcn_global_load_async_to_lds_b128(
      (__attribute__((address_space(1))) v4i*)(uintptr_t)g,
      (__attribute__((address_space(3))) v4i*)(uint32_t)(uintptr_t)l,
      0, 0);
#else
  *(uint4*)l = *(const uint4*)g;
#endif
}

// ---------------------------------------------------------------- small front-end

__global__ void k_deg(const int* __restrict__ dst, float* __restrict__ deg) {
  int e = blockIdx.x * 256 + threadIdx.x;
  if (e < NE) atomicAdd(&deg[dst[e]], 1.0f);
}

__global__ void k_dinv(const float* __restrict__ deg, float* __restrict__ dinv) {
  int i = blockIdx.x * 256 + threadIdx.x;
  if (i < N_NODES) dinv[i] = rsqrtf(deg[i] + 1.0f);
}

__global__ void k_embed(const int* __restrict__ ids, const float* __restrict__ emb,
                        float* __restrict__ x) {
  int gid = blockIdx.x * 256 + threadIdx.x;
  if (gid < N_NODES * H) {
    int i = gid >> 5, c = gid & 31;
    x[gid] = emb[ids[i] * H + c];
  }
}

// h = x @ W  (N x 32) @ (32 x 32)
__global__ void k_mm(const float* __restrict__ x, const float* __restrict__ W,
                     float* __restrict__ h) {
  int gid = blockIdx.x * 256 + threadIdx.x;
  if (gid < N_NODES * H) {
    int i = gid >> 5, o = gid & 31;
    float acc = 0.f;
#pragma unroll
    for (int c = 0; c < H; ++c) acc += x[i * H + c] * W[c * H + o];
    h[gid] = acc;
  }
}

// agg = h * dinv^2  (self term; also initializes agg)
__global__ void k_aggself(const float* __restrict__ h, const float* __restrict__ dinv,
                          float* __restrict__ agg) {
  int gid = blockIdx.x * 256 + threadIdx.x;
  if (gid < N_NODES * H) {
    int i = gid >> 5;
    agg[gid] = h[gid] * dinv[i] * dinv[i];
  }
}

__global__ void k_aggedge(const int* __restrict__ src, const int* __restrict__ dst,
                          const float* __restrict__ h, const float* __restrict__ dinv,
                          float* __restrict__ agg) {
  int gid = blockIdx.x * 256 + threadIdx.x;
  if (gid < NE * H) {
    int e = gid >> 5, c = gid & 31;
    int s = src[e], d = dst[e];
    atomicAdd(&agg[d * H + c], h[s * H + c] * dinv[s] * dinv[d]);
  }
}

// graphnorm over nodes (per channel) + relu; input = agg + gcn_bias
__global__ __launch_bounds__(256) void k_gnorm_relu(
    const float* __restrict__ agg, const float* __restrict__ gb,
    const float* __restrict__ g, const float* __restrict__ bt,
    const float* __restrict__ a, float* __restrict__ out) {
  __shared__ float red[256];
  __shared__ float smu, srs;
  const int c = blockIdx.x, tx = threadIdx.x;
  const float bias = gb[c], av = a[c];
  float s = 0.f;
  for (int i = tx; i < N_NODES; i += 256) s += agg[i * H + c] + bias;
  red[tx] = s; __syncthreads();
  for (int w = 128; w > 0; w >>= 1) { if (tx < w) red[tx] += red[tx + w]; __syncthreads(); }
  if (tx == 0) smu = red[0] / (float)N_NODES;
  __syncthreads();
  const float mu = smu;
  float s2 = 0.f;
  for (int i = tx; i < N_NODES; i += 256) {
    float v = agg[i * H + c] + bias - av * mu; s2 += v * v;
  }
  red[tx] = s2; __syncthreads();
  for (int w = 128; w > 0; w >>= 1) { if (tx < w) red[tx] += red[tx + w]; __syncthreads(); }
  if (tx == 0) srs = rsqrtf(red[0] / (float)N_NODES + EPSV);
  __syncthreads();
  const float rs = srs, gv = g[c], bv = bt[c];
  for (int i = tx; i < N_NODES; i += 256) {
    float v = agg[i * H + c] + bias - av * mu;
    out[i * H + c] = fmaxf(gv * v * rs + bv, 0.f);
  }
}

__global__ void k_eim(const int* __restrict__ src, const int* __restrict__ dst,
                      unsigned char* __restrict__ eim, unsigned char* __restrict__ eimT) {
  int e = blockIdx.x * 256 + threadIdx.x;
  if (e < NE) {
    int s = src[e], d = dst[e];
    eim [(size_t)s * N_NODES + d] = 1;
    eimT[(size_t)d * N_NODES + s] = 1;
  }
}

// ---------------------------------------------------------------- pair MLP (x1 / x2)
// Serves both: x1 -> (em=eim,  W=w_m1, out=X1 [c][i][j])
//              x2 -> (em=eimT, W=w_m2, out=X2T [c][k][j])  (pair is symmetric)
__global__ __launch_bounds__(256) void k_pair_mlp(
    const float* __restrict__ xn, const unsigned char* __restrict__ em,
    const float* __restrict__ w, const float* __restrict__ bias,
    uint16_t* __restrict__ Xo) {
  __shared__ float sw[33 * H];
  __shared__ float sb[H];
  __shared__ float sxr[H];
  const int tx = threadIdx.x;
  const int r = blockIdx.y;
  const int qc = blockIdx.x * 256 + tx;
  for (int tt = tx; tt < 33 * H; tt += 256) sw[tt] = w[tt];
  if (tx < H) { sb[tx] = bias[tx]; sxr[tx] = xn[r * H + tx]; }
  __syncthreads();
  float pair[H];
#pragma unroll
  for (int c = 0; c < H; ++c) pair[c] = sxr[c] * xn[qc * H + c];
  const float e = (float)em[(size_t)r * N_NODES + qc];
  for (int h = 0; h < H; ++h) {
    float acc = sb[h] + e * sw[32 * H + h];
#pragma unroll
    for (int c = 0; c < H; ++c) acc += pair[c] * sw[c * H + h];
    acc = fmaxf(acc, 0.f);
    Xo[(size_t)h * NN + (size_t)r * N_NODES + qc] = f2bf(acc);
  }
}

// ---------------------------------------------------------------- batched WMMA GEMM
// P[c] = X1[c] (1024x1024) @ X2T[c]^T ; X2T stored [c][n_out][k] so B-frags are K-contiguous.
// Double-buffered LDS; async global->LDS (ASYNCcnt) when available: one barrier +
// one s_wait_asynccnt per K-step, next tile streams in while WMMAs issue.
#define BM 128
#define BN 128
#define BK 32
#define LDK 40            // padded halves per LDS row (20 words -> conflict-free b128 frag loads)
#define TILE_H (BM * LDK) // halves per buffer per matrix

__global__ __launch_bounds__(256) void k_gemm(
    const uint16_t* __restrict__ X1, const uint16_t* __restrict__ X2T,
    float* __restrict__ P) {
  __shared__ __align__(16) uint16_t As[2 * TILE_H];
  __shared__ __align__(16) uint16_t Bs[2 * TILE_H];
  const int c  = blockIdx.z;
  const int i0 = blockIdx.y * BM;
  const int n0 = blockIdx.x * BN;
  const uint16_t* A = X1  + (size_t)c * NN;
  const uint16_t* B = X2T + (size_t)c * NN;
  float* Pp = P + (size_t)c * NN;

  const int t = threadIdx.x;
  const int lane = t & 31;
  const int wave = t >> 5;   // 8 waves
  const int wm = wave >> 1;  // 0..3 : 32-row group
  const int wn = wave & 1;   // 0..1 : 64-col group

  f32x8 acc[2][4];
#pragma unroll
  for (int mt = 0; mt < 2; ++mt)
#pragma unroll
    for (int nt = 0; nt < 4; ++nt)
#pragma unroll
      for (int v = 0; v < 8; ++v) acc[mt][nt][v] = 0.0f;

  // tile staging: 512 16-byte chunks per tile, 2 per thread per matrix
  const int r0 = t >> 2,         o0 = (t & 3) * 8;
  const int r1 = (t + 256) >> 2, o1 = ((t + 256) & 3) * 8;

  auto stage = [&](int kt, int buf) {
    uint16_t* al = &As[buf * TILE_H];
    uint16_t* bl = &Bs[buf * TILE_H];
    stage_chunk(&A[(size_t)(i0 + r0) * N_NODES + kt + o0], &al[r0 * LDK + o0]);
    stage_chunk(&A[(size_t)(i0 + r1) * N_NODES + kt + o1], &al[r1 * LDK + o1]);
    stage_chunk(&B[(size_t)(n0 + r0) * N_NODES + kt + o0], &bl[r0 * LDK + o0]);
    stage_chunk(&B[(size_t)(n0 + r1) * N_NODES + kt + o1], &bl[r1 * LDK + o1]);
#if !HAS_ASYNC_LDS
    if (kt + BK < N_NODES) {  // CDNA5 global_prefetch_b8 for the tile after next
      __builtin_prefetch(&A[(size_t)(i0 + r0) * N_NODES + kt + BK + o0], 0, 3);
      __builtin_prefetch(&B[(size_t)(n0 + r0) * N_NODES + kt + BK + o0], 0, 3);
    }
#endif
  };

  stage(0, 0);
  const int NT = N_NODES / BK;
  for (int n = 0; n < NT; ++n) {
#if HAS_ASYNC_LDS
    __builtin_amdgcn_s_wait_asynccnt(0);   // my tile-n async writes landed in LDS
#endif
    __syncthreads();                        // all waves: tile n visible; prev compute done
    if (n + 1 < NT) stage((n + 1) * BK, (n + 1) & 1);  // overlaps with WMMAs below

    const uint16_t* al = &As[(n & 1) * TILE_H];
    const uint16_t* bl = &Bs[(n & 1) * TILE_H];

    // A fragment: lanes 0-15 row M, halves K = kb..kb+7 then kb+16..kb+23 (kb = 8*(lane>>4))
    FragAB fa[2], fb[4];
    const int am = wm * 32 + (lane & 15);
    const int kb = (lane >> 4) * 8;
#pragma unroll
    for (int mt = 0; mt < 2; ++mt) {
      const int r = am + mt * 16;
      fa[mt].q[0] = *(const uint4*)&al[r * LDK + kb];
      fa[mt].q[1] = *(const uint4*)&al[r * LDK + kb + 16];
    }
    // B fragment: lanes 0-15 col N, halves K = kbb..kbb+7 then kbb+16..kbb+23 (kbb = 8*(lane>>4))
    const int bn = wn * 64 + (lane & 15);
    const int kbb = (lane >> 4) * 8;
#pragma unroll
    for (int nt = 0; nt < 4; ++nt) {
      const int r = bn + nt * 16;
      fb[nt].q[0] = *(const uint4*)&bl[r * LDK + kbb];
      fb[nt].q[1] = *(const uint4*)&bl[r * LDK + kbb + 16];
    }
#pragma unroll
    for (int mt = 0; mt < 2; ++mt)
#pragma unroll
      for (int nt = 0; nt < 4; ++nt)
        acc[mt][nt] = __builtin_amdgcn_wmma_f32_16x16x32_bf16(
            false, fa[mt].v, false, fb[nt].v, (short)0, acc[mt][nt], false, false);
  }

  // C/D layout: VGPR v -> M = v + 8*(lane>>4), N = lane&15
  const int mh = (lane >> 4) * 8;
  const int nc0 = lane & 15;
#pragma unroll
  for (int mt = 0; mt < 2; ++mt) {
    const int mb = i0 + wm * 32 + mt * 16 + mh;
#pragma unroll
    for (int nt = 0; nt < 4; ++nt) {
      const int nc = n0 + wn * 64 + nt * 16 + nc0;
#pragma unroll
      for (int v = 0; v < 8; ++v)
        Pp[(size_t)(mb + v) * N_NODES + nc] = acc[mt][nt][v];
    }
  }
}

// ---------------------------------------------------------------- graphnorm stats over (i,j)
__global__ __launch_bounds__(256) void k_stats(
    const float* __restrict__ xn, const unsigned char* __restrict__ eim,
    const float* __restrict__ P, const float* __restrict__ w3,
    const float* __restrict__ b3, float* __restrict__ gstats) {
  __shared__ float sw[65 * H];
  __shared__ float sb[H];
  __shared__ float sxi[H];
  __shared__ float sacc[2 * H];
  const int tx = threadIdx.x;
  const int i = blockIdx.x;
  for (int tt = tx; tt < 65 * H; tt += 256) sw[tt] = w3[tt];
  if (tx < H) { sb[tx] = b3[tx]; sxi[tx] = xn[i * H + tx]; }
  if (tx < 2 * H) sacc[tx] = 0.f;
  __syncthreads();
  float zs[H], zq[H];
#pragma unroll
  for (int h = 0; h < H; ++h) { zs[h] = 0.f; zq[h] = 0.f; }
  for (int j = tx; j < N_NODES; j += 256) {
    float pair[H], pv[H];
#pragma unroll
    for (int c = 0; c < H; ++c) pair[c] = sxi[c] * xn[j * H + c];
#pragma unroll
    for (int c = 0; c < H; ++c) pv[c] = P[(size_t)c * NN + (size_t)i * N_NODES + j];
    const float e = (float)eim[(size_t)i * N_NODES + j];
    for (int h = 0; h < H; ++h) {
      float z = sb[h] + e * sw[32 * H + h];
#pragma unroll
      for (int c = 0; c < H; ++c) z += pair[c] * sw[c * H + h];
#pragma unroll
      for (int c = 0; c < H; ++c) z += pv[c] * sw[(33 + c) * H + h];
      zs[h] += z; zq[h] += z * z;
    }
  }
  for (int h = 0; h < H; ++h) {
    atomicAdd(&sacc[h], zs[h]);
    atomicAdd(&sacc[H + h], zq[h]);
  }
  __syncthreads();
  if (tx < 2 * H) atomicAdd(&gstats[tx], sacc[tx]);
}

__global__ void k_finstats(const float* __restrict__ st, const float* __restrict__ a,
                           float* __restrict__ musig) {
  int h = threadIdx.x;
  if (h < H) {
    float mu = st[h] / (float)NN;
    float e2 = st[H + h] / (float)NN;
    float av = a[h];
    float var = e2 - 2.f * av * mu * mu + av * av * mu * mu;   // E[(z - a*mu)^2]
    musig[h] = mu;
    musig[H + h] = rsqrtf(var + EPSV);
  }
}

// ---------------------------------------------------------------- gather + output
__global__ __launch_bounds__(256) void k_out(
    const int* __restrict__ pos, const float* __restrict__ xn,
    const unsigned char* __restrict__ eim, const float* __restrict__ P,
    const float* __restrict__ w3, const float* __restrict__ b3,
    const float* __restrict__ g3, const float* __restrict__ bt3,
    const float* __restrict__ a3, const float* __restrict__ musig,
    const float* __restrict__ wd, const float* __restrict__ bd,
    float* __restrict__ out) {
  __shared__ float sw[65 * H];
  __shared__ float sb[H], sg[H], sbt[H], sa[H], smu[H], srs[H], swd[H];
  const int tx = threadIdx.x;
  for (int tt = tx; tt < 65 * H; tt += 256) sw[tt] = w3[tt];
  if (tx < H) {
    sb[tx] = b3[tx]; sg[tx] = g3[tx]; sbt[tx] = bt3[tx]; sa[tx] = a3[tx];
    smu[tx] = musig[tx]; srs[tx] = musig[H + tx]; swd[tx] = wd[tx];
  }
  __syncthreads();
  const int q = blockIdx.x * 256 + tx;
  if (q >= NQ) return;
  const int i = pos[2 * q], j = pos[2 * q + 1];
  float pair[H], pij[H], pji[H];
#pragma unroll
  for (int c = 0; c < H; ++c) pair[c] = xn[i * H + c] * xn[j * H + c];
#pragma unroll
  for (int c = 0; c < H; ++c) {
    pij[c] = P[(size_t)c * NN + (size_t)i * N_NODES + j];
    pji[c] = P[(size_t)c * NN + (size_t)j * N_NODES + i];
  }
  const float eij = (float)eim[(size_t)i * N_NODES + j];
  const float eji = (float)eim[(size_t)j * N_NODES + i];
  float res = 0.f;
  for (int h = 0; h < H; ++h) {
    float zij = sb[h] + eij * sw[32 * H + h];
    float zji = sb[h] + eji * sw[32 * H + h];
#pragma unroll
    for (int c = 0; c < H; ++c) {
      const float w = sw[c * H + h];
      zij += pair[c] * w; zji += pair[c] * w;
    }
#pragma unroll
    for (int c = 0; c < H; ++c) {
      const float w = sw[(33 + c) * H + h];
      zij += pij[c] * w; zji += pji[c] * w;
    }
    const float yij = fmaxf(sg[h] * (zij - sa[h] * smu[h]) * srs[h] + sbt[h], 0.f);
    const float yji = fmaxf(sg[h] * (zji - sa[h] * smu[h]) * srs[h] + sbt[h], 0.f);
    res += yij * yji * swd[h];
  }
  out[q] = res + bd[0];
}

// ---------------------------------------------------------------- launch

extern "C" void kernel_launch(void* const* d_in, const int* in_sizes, int n_in,
                              void* d_out, int out_size, void* d_ws, size_t ws_size,
                              hipStream_t stream) {
  (void)in_sizes; (void)n_in; (void)out_size; (void)ws_size;
  const int*   x_ids  = (const int*)  d_in[0];
  const int*   ei     = (const int*)  d_in[1];   // [2][E]: src = ei, dst = ei+NE
  const int*   pos    = (const int*)  d_in[2];
  const float* emb    = (const float*)d_in[3];
  const float* gcn_w0 = (const float*)d_in[4];
  const float* gcn_b0 = (const float*)d_in[5];
  const float* gn0_g  = (const float*)d_in[6];
  const float* gn0_b  = (const float*)d_in[7];
  const float* gn0_a  = (const float*)d_in[8];
  const float* gcn_w1 = (const float*)d_in[9];
  const float* gcn_b1 = (const float*)d_in[10];
  const float* gn1_g  = (const float*)d_in[11];
  const float* gn1_b  = (const float*)d_in[12];
  const float* gn1_a  = (const float*)d_in[13];
  const float* w_m1   = (const float*)d_in[14];
  const float* b_m1   = (const float*)d_in[15];
  const float* w_m2   = (const float*)d_in[16];
  const float* b_m2   = (const float*)d_in[17];
  const float* w_m3   = (const float*)d_in[18];
  const float* b_m3   = (const float*)d_in[19];
  const float* gn3_g  = (const float*)d_in[20];
  const float* gn3_b  = (const float*)d_in[21];
  const float* gn3_a  = (const float*)d_in[22];
  const float* w_dir  = (const float*)d_in[23];
  const float* b_dir  = (const float*)d_in[24];
  float* out = (float*)d_out;

  char* base = (char*)d_ws;
  size_t off = 0;
  auto alloc = [&](size_t bytes) -> void* {
    void* r = base + off;
    off = (off + bytes + 255) & ~(size_t)255;
    return r;
  };
  float*         deg   = (float*)        alloc((size_t)N_NODES * 4);
  float*         dinv  = (float*)        alloc((size_t)N_NODES * 4);
  float*         x0    = (float*)        alloc((size_t)N_NODES * H * 4);
  float*         hbuf  = (float*)        alloc((size_t)N_NODES * H * 4);
  float*         agg   = (float*)        alloc((size_t)N_NODES * H * 4);
  unsigned char* eim   = (unsigned char*)alloc((size_t)NN);
  unsigned char* eimT  = (unsigned char*)alloc((size_t)NN);
  uint16_t*      X1    = (uint16_t*)     alloc((size_t)H * NN * 2);
  uint16_t*      X2T   = (uint16_t*)     alloc((size_t)H * NN * 2);
  float*         P     = (float*)        alloc((size_t)H * NN * 4);
  float*         gst   = (float*)        alloc(2 * H * 4);
  float*         musig = (float*)        alloc(2 * H * 4);

  const int* src = ei;
  const int* dst = ei + NE;

  (void)hipMemsetAsync(deg, 0, (size_t)N_NODES * 4, stream);
  k_deg  <<<NE / 256, 256, 0, stream>>>(dst, deg);
  k_dinv <<<N_NODES / 256, 256, 0, stream>>>(deg, dinv);
  k_embed<<<(N_NODES * H) / 256, 256, 0, stream>>>(x_ids, emb, x0);

  // GCN layer 0
  k_mm     <<<(N_NODES * H) / 256, 256, 0, stream>>>(x0, gcn_w0, hbuf);
  k_aggself<<<(N_NODES * H) / 256, 256, 0, stream>>>(hbuf, dinv, agg);
  k_aggedge<<<(NE * H) / 256, 256, 0, stream>>>(src, dst, hbuf, dinv, agg);
  k_gnorm_relu<<<H, 256, 0, stream>>>(agg, gcn_b0, gn0_g, gn0_b, gn0_a, x0);
  // GCN layer 1
  k_mm     <<<(N_NODES * H) / 256, 256, 0, stream>>>(x0, gcn_w1, hbuf);
  k_aggself<<<(N_NODES * H) / 256, 256, 0, stream>>>(hbuf, dinv, agg);
  k_aggedge<<<(NE * H) / 256, 256, 0, stream>>>(src, dst, hbuf, dinv, agg);
  k_gnorm_relu<<<H, 256, 0, stream>>>(agg, gcn_b1, gn1_g, gn1_b, gn1_a, x0);

  (void)hipMemsetAsync(eim,  0, (size_t)NN, stream);
  (void)hipMemsetAsync(eimT, 0, (size_t)NN, stream);
  k_eim<<<NE / 256, 256, 0, stream>>>(src, dst, eim, eimT);

  dim3 pg(N_NODES / 256, N_NODES);
  k_pair_mlp<<<pg, 256, 0, stream>>>(x0, eim,  w_m1, b_m1, X1);
  k_pair_mlp<<<pg, 256, 0, stream>>>(x0, eimT, w_m2, b_m2, X2T);

  dim3 gg(N_NODES / BN, N_NODES / BM, H);
  k_gemm<<<gg, 256, 0, stream>>>(X1, X2T, P);

  (void)hipMemsetAsync(gst, 0, 2 * H * 4, stream);
  k_stats<<<N_NODES, 256, 0, stream>>>(x0, eim, P, w_m3, b_m3, gst);
  k_finstats<<<1, 64, 0, stream>>>(gst, gn3_a, musig);
  k_out<<<NQ / 256, 256, 0, stream>>>(pos, x0, eim, P, w_m3, b_m3,
                                      gn3_g, gn3_b, gn3_a, musig, w_dir, b_dir, out);
}
